// Attention_87462714015865
// MI455X (gfx1250) — compile-verified, matrix operand order checked by reading, not run on hardware
//
#include <hip/hip_runtime.h>
#include <hip/hip_bf16.h>

// ---------------------------------------------------------------------------
// Problem constants (from reference): B=2, L=2048, D=2048, H=16, Hd=128
// ---------------------------------------------------------------------------
#define BB 2
#define LL 2048
#define DD 2048
#define HH 16
#define HD 128
#define ML (BB * LL)          // 4096 rows of the "token" matrices

typedef __attribute__((ext_vector_type(16))) __bf16 v16bf;
typedef __attribute__((ext_vector_type(8)))  __bf16 bf16x8;
typedef __attribute__((ext_vector_type(8)))  float  v8f;

struct Frag {
    union { v16bf v; bf16x8 h[2]; };
};

static __device__ inline v8f wmma_bf16(v16bf a, v16bf b, v8f c) {
    // D = A(16x32 bf16) * B(32x16 bf16) + C(16x16 f32)
    return __builtin_amdgcn_wmma_f32_16x16x32_bf16(
        /*neg_a=*/false, a, /*neg_b=*/false, b,
        /*c_mod=*/(short)0, c, /*reuse_a=*/false, /*reuse_b=*/false);
}

static __device__ inline v8f v8f_zero() {
    v8f v;
#pragma unroll
    for (int i = 0; i < 8; ++i) v[i] = 0.0f;
    return v;
}

// ---------------------------------------------------------------------------
// Async global->LDS staging (CDNA5 ASYNCcnt path).
//   GVS mode: LDS address from the vdst VGPR; global address = SGPR base + VGPR byte offset
//   Low 32 bits of a generic pointer to __shared__ are the LDS byte offset.
// ---------------------------------------------------------------------------
static __device__ inline void async_ld_b128(uint32_t lds_off, uint32_t byte_off,
                                            uint64_t base) {
    asm volatile("global_load_async_to_lds_b128 %0, %1, %2"
                 :: "v"(lds_off), "v"(byte_off), "s"(base) : "memory");
}
static __device__ inline void s_wait_async0() {
    asm volatile("s_wait_asynccnt 0" ::: "memory");
}

// ---------------------------------------------------------------------------
// Kernel 1: f32 -> bf16 cast (grid-stride)
// ---------------------------------------------------------------------------
__global__ void f32_to_bf16_kernel(const float* __restrict__ src,
                                   __bf16* __restrict__ dst, int n) {
    for (int i = blockIdx.x * blockDim.x + threadIdx.x; i < n;
         i += gridDim.x * blockDim.x) {
        dst[i] = (__bf16)src[i];
    }
}

// ---------------------------------------------------------------------------
// Kernel 2: generic bf16 GEMM   C[m][n] = sum_k A[m][k] * W[n][k]
//   A: MxK bf16 row-major, W: NxK bf16 row-major (i.e. C = A @ W^T)
//   Block tile 128x128, 8 waves (2x4), wave tile 64x32, K-step 32.
//   Double-buffered LDS, filled by async global->LDS loads.
// ---------------------------------------------------------------------------
__global__ __launch_bounds__(256) void gemm_bf16_kernel(
    const __bf16* __restrict__ A, const __bf16* __restrict__ W,
    float* __restrict__ Cf, __bf16* __restrict__ Cb,
    int M, int N, int Kd, int outBf) {
    __shared__ __bf16 lA[2][128 * 40];   // row stride 40 (pad) for bank spread
    __shared__ __bf16 lB[2][128 * 40];

    const int t    = threadIdx.x;
    const int lane = t & 31;
    const int wid  = t >> 5;          // 0..7
    const int wm   = wid >> 2;        // 0..1  (M direction)
    const int wn   = wid & 3;         // 0..3  (N direction)
    const int bM   = blockIdx.y * 128;
    const int bN   = blockIdx.x * 128;
    const int kh   = lane >> 4;       // half-wave index
    const int lr   = lane & 15;

    const uint64_t baseA = (uint64_t)(uintptr_t)A;
    const uint64_t baseW = (uint64_t)(uintptr_t)W;
    const int r0 = t >> 2;            // 0..63
    const int c0 = t & 3;             // 0..3  (8-element chunk within 32)

    // Each thread stages rows r0 and r0+64, chunk c0, for both tiles.
    auto issue = [&](int buf, int k0) {
#pragma unroll
        for (int i = 0; i < 2; ++i) {
            int r = r0 + i * 64;
            uint32_t lofs = (uint32_t)(r * 40 + c0 * 8);
            async_ld_b128((uint32_t)(uintptr_t)&lA[buf][lofs],
                          (uint32_t)(((size_t)(bM + r) * Kd + k0 + c0 * 8) * 2),
                          baseA);
            async_ld_b128((uint32_t)(uintptr_t)&lB[buf][lofs],
                          (uint32_t)(((size_t)(bN + r) * Kd + k0 + c0 * 8) * 2),
                          baseW);
        }
    };

    v8f acc[4][2];
#pragma unroll
    for (int i = 0; i < 4; ++i)
#pragma unroll
        for (int j = 0; j < 2; ++j) acc[i][j] = v8f_zero();

    int buf = 0;
    issue(0, 0);
    s_wait_async0();
    __syncthreads();

    for (int k0 = 0; k0 < Kd; k0 += 32) {
        if (k0 + 32 < Kd) issue(buf ^ 1, k0 + 32);   // prefetch next tile

        Frag af[4], bf[2];
#pragma unroll
        for (int i = 0; i < 4; ++i) {
            const __bf16* p = &lA[buf][(wm * 64 + i * 16 + lr) * 40 + kh * 8];
            af[i].h[0] = *(const bf16x8*)p;
            af[i].h[1] = *(const bf16x8*)(p + 16);
        }
#pragma unroll
        for (int j = 0; j < 2; ++j) {
            const __bf16* p = &lB[buf][(wn * 32 + j * 16 + lr) * 40 + kh * 8];
            bf[j].h[0] = *(const bf16x8*)p;
            bf[j].h[1] = *(const bf16x8*)(p + 16);
        }
#pragma unroll
        for (int i = 0; i < 4; ++i)
#pragma unroll
            for (int j = 0; j < 2; ++j)
                acc[i][j] = wmma_bf16(af[i].v, bf[j].v, acc[i][j]);

        s_wait_async0();      // prefetched tile has landed
        __syncthreads();      // all waves done reading buf / writing buf^1
        buf ^= 1;
    }

    // Epilogue: D layout = lane holds N=lr, rows M = kh*8 + r (VGPR r)
#pragma unroll
    for (int i = 0; i < 4; ++i) {
#pragma unroll
        for (int j = 0; j < 2; ++j) {
            int m0 = bM + wm * 64 + i * 16 + kh * 8;
            int n  = bN + wn * 32 + j * 16 + lr;
#pragma unroll
            for (int r = 0; r < 8; ++r) {
                size_t off = (size_t)(m0 + r) * N + n;
                if (outBf) Cb[off] = (__bf16)acc[i][j][r];
                else       Cf[off] = acc[i][j][r];
            }
        }
    }
}

// ---------------------------------------------------------------------------
// Kernel 3: RoPE on Q,K + relayout [b,l,h,d]->[b,h,l,d]; V -> [b,h,d,l]
//   idx encodes (b, h, l, d<64): d:6 bits, l:11, h:4, b:1  => 2^22 items
// ---------------------------------------------------------------------------
__global__ void rope_relayout_kernel(const __bf16* __restrict__ Qt,
                                     const __bf16* __restrict__ Kt,
                                     const __bf16* __restrict__ Vt,
                                     const float* __restrict__ cosb,
                                     const float* __restrict__ sinb,
                                     __bf16* __restrict__ Qo,
                                     __bf16* __restrict__ Ko,
                                     __bf16* __restrict__ Vo) {
    const int total = BB * HH * LL * (HD / 2);
    for (int idx = blockIdx.x * blockDim.x + threadIdx.x; idx < total;
         idx += gridDim.x * blockDim.x) {
        int d = idx & 63;
        int l = (idx >> 6) & (LL - 1);
        int h = (idx >> 17) & (HH - 1);
        int b = idx >> 21;

        size_t src = ((size_t)(b * LL + l)) * DD + h * HD + d;
        float c0 = cosb[l * HD + d],      s0 = sinb[l * HD + d];
        float c1 = cosb[l * HD + d + 64], s1 = sinb[l * HD + d + 64];

        float q0 = (float)Qt[src], q1 = (float)Qt[src + 64];
        float k0 = (float)Kt[src], k1 = (float)Kt[src + 64];

        size_t dst = (((size_t)(b * HH + h)) * LL + l) * HD + d;
        Qo[dst]      = (__bf16)(q0 * c0 - q1 * s0);
        Qo[dst + 64] = (__bf16)(q1 * c1 + q0 * s1);
        Ko[dst]      = (__bf16)(k0 * c0 - k1 * s0);
        Ko[dst + 64] = (__bf16)(k1 * c1 + k0 * s1);

        // V: transpose within head -> [b][h][d][l]
        size_t vdst = (((size_t)(b * HH + h)) * HD + d) * LL + l;
        Vo[vdst]                   = Vt[src];
        Vo[vdst + (size_t)64 * LL] = Vt[src + 64];
    }
}

// ---------------------------------------------------------------------------
// Kernel 4: flash attention.
//   grid (L/64, H, B); block 128 = 4 waves; each wave owns 16 q rows.
//   Q,K: [b][h][l][128] bf16;  Vt: [b][h][128][l] bf16;  AO: [b*l][2048] bf16
//   K/V chunks double-buffered in LDS via async global->LDS loads.
// ---------------------------------------------------------------------------
__global__ __launch_bounds__(128) void flash_attn_kernel(
    const __bf16* __restrict__ Q, const __bf16* __restrict__ K,
    const __bf16* __restrict__ Vt, __bf16* __restrict__ AO) {
    __shared__ __bf16 lK[2][32 * 136];   // 32 keys x 128 d   (pad 136)
    __shared__ __bf16 lV[2][128 * 40];   // 128 d x 32 keys   (pad 40)
    __shared__ __bf16 lP[4][16 * 40];    // per-wave P transpose staging

    const int t    = threadIdx.x;
    const int lane = t & 31;
    const int wid  = t >> 5;
    const int kh   = lane >> 4;
    const int lr   = lane & 15;

    const int b  = blockIdx.z;
    const int h  = blockIdx.y;
    const int q0 = blockIdx.x * 64;
    const int qrow = q0 + wid * 16;

    const size_t base  = ((size_t)(b * HH + h)) * LL * HD;  // Q/K [l][d]
    const size_t vbase = ((size_t)(b * HH + h)) * HD * LL;  // Vt  [d][l]
    const uint64_t baseK = (uint64_t)(uintptr_t)K;
    const uint64_t baseV = (uint64_t)(uintptr_t)Vt;

    auto issueKV = [&](int buf, int kc) {
#pragma unroll
        for (int i = 0; i < 4; ++i) {
            int idx = t + i * 128;
            int rk = idx >> 4, ck = idx & 15;
            async_ld_b128((uint32_t)(uintptr_t)&lK[buf][rk * 136 + ck * 8],
                          (uint32_t)((base + (size_t)(kc + rk) * HD + ck * 8) * 2),
                          baseK);
            int rv = idx >> 2, cv = idx & 3;
            async_ld_b128((uint32_t)(uintptr_t)&lV[buf][rv * 40 + cv * 8],
                          (uint32_t)((vbase + (size_t)rv * LL + kc + cv * 8) * 2),
                          baseV);
        }
    };

    // Q fragments: 16 rows x 128 d -> 4 A-frags (K-chunks of 32)
    Frag qf[4];
#pragma unroll
    for (int c = 0; c < 4; ++c) {
        const __bf16* p = Q + base + (size_t)(qrow + lr) * HD + c * 32 + kh * 8;
        qf[c].h[0] = *(const bf16x8*)p;
        qf[c].h[1] = *(const bf16x8*)(p + 16);
    }

    v8f o[8];
#pragma unroll
    for (int i = 0; i < 8; ++i) o[i] = v8f_zero();
    float mrow[8], srow[8];
#pragma unroll
    for (int r = 0; r < 8; ++r) { mrow[r] = -1e30f; srow[r] = 0.0f; }

    const float sc = 0.08838834764831845f;  // 1/sqrt(128)

    int buf = 0;
    issueKV(0, 0);
    s_wait_async0();
    __syncthreads();

    for (int kc = 0; kc < LL; kc += 32) {
        if (kc + 32 < LL) issueKV(buf ^ 1, kc + 32);   // prefetch next chunk

        // S = Q K^T for two 16-key tiles (s0: keys kc..kc+15, s1: +16)
        v8f s0 = v8f_zero(), s1 = v8f_zero();
#pragma unroll
        for (int c = 0; c < 4; ++c) {
            Frag kf0, kf1;
            const __bf16* p0 = &lK[buf][lr * 136 + c * 32 + kh * 8];
            const __bf16* p1 = &lK[buf][(16 + lr) * 136 + c * 32 + kh * 8];
            kf0.h[0] = *(const bf16x8*)p0;  kf0.h[1] = *(const bf16x8*)(p0 + 16);
            kf1.h[0] = *(const bf16x8*)p1;  kf1.h[1] = *(const bf16x8*)(p1 + 16);
            s0 = wmma_bf16(qf[c].v, kf0.v, s0);
            s1 = wmma_bf16(qf[c].v, kf1.v, s1);
        }

        // Online softmax over the 32 new columns.
        float alpha[8];
#pragma unroll
        for (int r = 0; r < 8; ++r) {
            float a = s0[r] * sc, bq = s1[r] * sc;
            float mx = fmaxf(a, bq);
#pragma unroll
            for (int off = 1; off < 16; off <<= 1)
                mx = fmaxf(mx, __shfl_xor(mx, off, 32));
            float mnew = fmaxf(mrow[r], mx);
            float al = __expf(mrow[r] - mnew);
            float p0 = __expf(a - mnew);
            float p1 = __expf(bq - mnew);
            s0[r] = p0; s1[r] = p1;
            float ps = p0 + p1;
#pragma unroll
            for (int off = 1; off < 16; off <<= 1)
                ps += __shfl_xor(ps, off, 32);
            srow[r] = srow[r] * al + ps;
            mrow[r] = mnew;
            alpha[r] = al;
        }
#pragma unroll
        for (int dt = 0; dt < 8; ++dt)
#pragma unroll
            for (int r = 0; r < 8; ++r) o[dt][r] *= alpha[r];

        // Transpose P (D-layout -> A-layout) through per-wave LDS staging.
#pragma unroll
        for (int r = 0; r < 8; ++r) {
            lP[wid][(kh * 8 + r) * 40 + lr]      = (__bf16)s0[r];
            lP[wid][(kh * 8 + r) * 40 + lr + 16] = (__bf16)s1[r];
        }
        Frag pf;
        {
            const __bf16* p = &lP[wid][lr * 40 + kh * 8];
            pf.h[0] = *(const bf16x8*)p;
            pf.h[1] = *(const bf16x8*)(p + 16);
        }

        // O += P * V
#pragma unroll
        for (int dt = 0; dt < 8; ++dt) {
            Frag vf;
            const __bf16* p = &lV[buf][(dt * 16 + lr) * 40 + kh * 8];
            vf.h[0] = *(const bf16x8*)p;
            vf.h[1] = *(const bf16x8*)(p + 16);
            o[dt] = wmma_bf16(pf.v, vf.v, o[dt]);
        }

        s_wait_async0();      // prefetched K/V chunk has landed
        __syncthreads();
        buf ^= 1;
    }

    // Epilogue: normalize and write AO as [b*l][H*Hd] bf16 (GEMM-ready).
#pragma unroll
    for (int dt = 0; dt < 8; ++dt) {
#pragma unroll
        for (int r = 0; r < 8; ++r) {
            float val = o[dt][r] / srow[r];
            size_t off = ((size_t)(b * LL + qrow + kh * 8 + r)) * DD +
                         h * HD + dt * 16 + lr;
            AO[off] = (__bf16)val;
        }
    }
}

// ---------------------------------------------------------------------------
// Host-side orchestration
// ---------------------------------------------------------------------------
extern "C" void kernel_launch(void* const* d_in, const int* in_sizes, int n_in,
                              void* d_out, int out_size, void* d_ws, size_t ws_size,
                              hipStream_t stream) {
    const float* hs   = (const float*)d_in[0];   // (B,L,D)
    const float* cosb = (const float*)d_in[1];   // (1,1,L,Hd)
    const float* sinb = (const float*)d_in[2];
    const float* wq   = (const float*)d_in[3];   // (H*Hd, D)
    const float* wk   = (const float*)d_in[4];
    const float* wv   = (const float*)d_in[5];
    const float* wo   = (const float*)d_in[6];   // (D, H*Hd)
    float* out = (float*)d_out;

    char* ws = (char*)d_ws;
    const size_t SZ_X  = (size_t)ML * DD * 2;    // 16.78 MB bf16 token matrix
    const size_t SZ_W  = (size_t)DD * DD * 2;    //  8.39 MB bf16 weight

    __bf16* Xbf   = (__bf16*)(ws);
    __bf16* Wqbf  = (__bf16*)(ws + SZ_X);
    __bf16* Wkbf  = (__bf16*)(ws + SZ_X + SZ_W);
    __bf16* Wvbf  = (__bf16*)(ws + SZ_X + 2 * SZ_W);
    __bf16* Wobf  = (__bf16*)(ws + SZ_X + 3 * SZ_W);
    __bf16* Qtmp  = (__bf16*)(ws + SZ_X + 4 * SZ_W);             // [m][n] bf16
    __bf16* Ktmp  = (__bf16*)(ws + SZ_X + 4 * SZ_W + SZ_X);
    __bf16* Vtmp  = (__bf16*)(ws + SZ_X + 4 * SZ_W + 2 * SZ_X);
    __bf16* Qbf   = (__bf16*)(ws + SZ_X + 4 * SZ_W + 3 * SZ_X);  // [b][h][l][d]
    __bf16* Kbf   = (__bf16*)(ws + SZ_X + 4 * SZ_W + 4 * SZ_X);
    __bf16* Vtbf  = (__bf16*)(ws + SZ_X + 4 * SZ_W + 5 * SZ_X);  // [b][h][d][l]
    __bf16* AObf  = Qtmp;  // alias: Qtmp is dead once attention runs

    // 1) casts to bf16
    f32_to_bf16_kernel<<<1024, 256, 0, stream>>>(hs, Xbf, ML * DD);
    f32_to_bf16_kernel<<<1024, 256, 0, stream>>>(wq, Wqbf, DD * DD);
    f32_to_bf16_kernel<<<1024, 256, 0, stream>>>(wk, Wkbf, DD * DD);
    f32_to_bf16_kernel<<<1024, 256, 0, stream>>>(wv, Wvbf, DD * DD);
    f32_to_bf16_kernel<<<1024, 256, 0, stream>>>(wo, Wobf, DD * DD);

    // 2) QKV projections (bf16 out)
    dim3 ggrid(DD / 128, ML / 128);
    gemm_bf16_kernel<<<ggrid, 256, 0, stream>>>(Xbf, Wqbf, nullptr, Qtmp,
                                                ML, DD, DD, 1);
    gemm_bf16_kernel<<<ggrid, 256, 0, stream>>>(Xbf, Wkbf, nullptr, Ktmp,
                                                ML, DD, DD, 1);
    gemm_bf16_kernel<<<ggrid, 256, 0, stream>>>(Xbf, Wvbf, nullptr, Vtmp,
                                                ML, DD, DD, 1);

    // 3) RoPE + relayout (+ V transpose)
    rope_relayout_kernel<<<8192, 256, 0, stream>>>(Qtmp, Ktmp, Vtmp,
                                                   cosb, sinb,
                                                   Qbf, Kbf, Vtbf);

    // 4) flash attention
    dim3 agrid(LL / 64, HH, BB);
    flash_attn_kernel<<<agrid, 128, 0, stream>>>(Qbf, Kbf, Vtbf, AObf);

    // 5) output projection (f32 out -> d_out)
    gemm_bf16_kernel<<<ggrid, 256, 0, stream>>>(AObf, Wobf, out, nullptr,
                                                ML, DD, DD, 0);
}
